// Model_8203387536000
// MI455X (gfx1250) — matrix operand order repaired and checked by reading, not compiled, not run on hardware
//
#include <hip/hip_runtime.h>
#include <math.h>

typedef __attribute__((ext_vector_type(16))) __bf16 v16bf;
typedef __attribute__((ext_vector_type(8)))  __bf16 v8bf;
typedef __attribute__((ext_vector_type(8)))  float  v8f;

namespace {
constexpr int B_ = 4, N_ = 16384, M_ = 256;
constexpr int H_ = 384, HEADS_ = 8, DH_ = 8, G_ = 32, DEPTH_ = 12;
constexpr int MU_ = 96, INNER_ = 64, CAH_ = 4, CAI_ = 32, IN_ = 20;
}

static __device__ __forceinline__ __bf16 f2bf(float f) { return (__bf16)f; }
static __device__ __forceinline__ float siluf_(float x) { return x / (1.f + __expf(-x)); }
static __device__ __forceinline__ float geluf_(float x) { return 0.5f * x * (1.f + erff(x * 0.70710678118654752f)); }

// ---------------------------------------------------------------------------
// bf16 WMMA GEMM:  out = epi( A[rows][Ka](bf16) @ Wt[Ca][Ka]^T(bf16) + bias )
//   A  : bf16 row-major, pitch Ka (K padded to mult of 32, zero-filled)
//   Wt : bf16 TRANSPOSED weights [col][k], pitch Ka, Ca = mult of 64 (padded)
// epi 0: DstF = v ; epi 1: DstB = bf16(gelu(v)) ; epi 2: DstF = Resid + Gate*v
// Staging is pure b128 copies; fragments are contiguous 16B ds_load_b128.
// Block: 128 threads (4 waves), tile 64x64.
// ---------------------------------------------------------------------------
__global__ __launch_bounds__(128)
void gemm_bf16(const __bf16* __restrict__ A, const __bf16* __restrict__ Wt,
               const float* __restrict__ bias,
               float* __restrict__ DstF, __bf16* __restrict__ DstB,
               const float* __restrict__ Resid, const float* __restrict__ Gate,
               int Ka, int C, int epi, int gstride, int goff, int rowsPerB)
{
    __shared__ __bf16 lA [64][40];   // A tile  [row][k], 80B pitch
    __shared__ __bf16 lBt[64][40];   // W tile  [col][k], 80B pitch

    const int r0   = blockIdx.x * 64;
    const int c0   = blockIdx.y * 64;
    const int tid  = threadIdx.x;
    const int wv   = tid >> 5;
    const int lane = tid & 31;
    const int lhalf = lane >> 4;
    const int l16   = lane & 15;

    // staging: thread -> (row 0..31, 16B chunk 0..3), second row at +32
    const int sr = tid >> 2;
    const int sc = (tid & 3) * 8;                 // halves
    const __bf16* Ap = A  + (size_t)(r0 + sr) * Ka + sc;
    const __bf16* Wp = Wt + (size_t)(c0 + sr) * Ka + sc;
    const size_t rowStep = (size_t)32 * Ka;
    __bf16* lAs = &lA [sr][sc];
    __bf16* lBs = &lBt[sr][sc];

    v8f zero = {0.f,0.f,0.f,0.f,0.f,0.f,0.f,0.f};
    v8f acc[4] = {zero, zero, zero, zero};

    for (int k0 = 0; k0 < Ka; k0 += 32) {
        v8bf a0 = *reinterpret_cast<const v8bf*>(Ap);
        v8bf a1 = *reinterpret_cast<const v8bf*>(Ap + rowStep);
        v8bf w0 = *reinterpret_cast<const v8bf*>(Wp);
        v8bf w1 = *reinterpret_cast<const v8bf*>(Wp + rowStep);
        *reinterpret_cast<v8bf*>(lAs)            = a0;
        *reinterpret_cast<v8bf*>(lAs + 32 * 40)  = a1;
        *reinterpret_cast<v8bf*>(lBs)            = w0;
        *reinterpret_cast<v8bf*>(lBs + 32 * 40)  = w1;
        Ap += 32; Wp += 32;
        __syncthreads();

        // A fragment: two contiguous 16B runs -> 2x ds_load_b128
        const __bf16* arow = &lA[wv * 16 + l16][0];
        v8bf f0 = *reinterpret_cast<const v8bf*>(arow + lhalf * 8);
        v8bf f1 = *reinterpret_cast<const v8bf*>(arow + 16 + lhalf * 8);
        v16bf af = __builtin_shufflevector(f0, f1,
            0,1,2,3,4,5,6,7,8,9,10,11,12,13,14,15);

        #pragma unroll
        for (int ct = 0; ct < 4; ++ct) {
            const __bf16* bcol = &lBt[ct * 16 + l16][8 * lhalf];
            v8bf b0 = *reinterpret_cast<const v8bf*>(bcol);
            v8bf b1 = *reinterpret_cast<const v8bf*>(bcol + 16);
            v16bf bf = __builtin_shufflevector(b0, b1,
                0,1,2,3,4,5,6,7,8,9,10,11,12,13,14,15);
            acc[ct] = __builtin_amdgcn_wmma_f32_16x16x32_bf16(
                false, af, false, bf, (short)0, acc[ct], false, false);
        }
        __syncthreads();
    }

    #pragma unroll
    for (int ct = 0; ct < 4; ++ct) {
        int col = c0 + ct * 16 + l16;
        if (col >= C) continue;
        float bv = bias ? bias[col] : 0.f;
        #pragma unroll
        for (int r = 0; r < 8; ++r) {
            int row = r0 + wv * 16 + r + lhalf * 8;
            float v = acc[ct][r] + bv;
            size_t idx = (size_t)row * C + col;
            if (epi == 1) {
                DstB[idx] = f2bf(geluf_(v));
            } else if (epi == 2) {
                int bb = row / rowsPerB;
                DstF[idx] = Resid[idx] + Gate[(size_t)bb * gstride + goff + col] * v;
            } else {
                DstF[idx] = v;
            }
        }
    }
}

// Weight convert: Wt[c][k] = bf16(W[k][C=c]) with zero padding to Ka x Ca
__global__ void wconvT_kernel(const float* __restrict__ W, __bf16* __restrict__ Wt,
                              int K, int C, int Ka, int n)
{
    int idx = blockIdx.x * 256 + threadIdx.x;
    if (idx >= n) return;
    int c = idx / Ka, k = idx - c * Ka;
    float v = (k < K && c < C) ? W[(size_t)k * C + c] : 0.f;
    Wt[idx] = f2bf(v);
}

// ---------------------------------------------------------------------------
// LayerNorm (+optional affine/addvec/adaLN mod); OutT = float or __bf16.
// One wave per row of 384; 8 rows per 256-thread block.
// ---------------------------------------------------------------------------
template<typename OutT>
__global__ __launch_bounds__(256)
void lnmod_kernel(const float* __restrict__ X, OutT* __restrict__ Y, int rows,
                  const float* __restrict__ g, const float* __restrict__ b,
                  const float* __restrict__ ada, int astride, int shoff, int scoff,
                  const float* __restrict__ addvec, int rowsPerB, float eps)
{
    const int wv = threadIdx.x >> 5, lane = threadIdx.x & 31;
    const int row = blockIdx.x * 8 + wv;
    if (row >= rows) return;
    const float* xr = X + (size_t)row * H_;
    float v[12];
    float s = 0.f;
    #pragma unroll
    for (int k = 0; k < 12; ++k) { v[k] = xr[lane + 32 * k]; s += v[k]; }
    #pragma unroll
    for (int off = 16; off > 0; off >>= 1) s += __shfl_xor(s, off, 32);
    float mean = s * (1.f / 384.f);
    float var = 0.f;
    #pragma unroll
    for (int k = 0; k < 12; ++k) { float d = v[k] - mean; var += d * d; }
    #pragma unroll
    for (int off = 16; off > 0; off >>= 1) var += __shfl_xor(var, off, 32);
    float inv = rsqrtf(var * (1.f / 384.f) + eps);
    int bb = row / rowsPerB;
    OutT* yr = Y + (size_t)row * H_;
    #pragma unroll
    for (int k = 0; k < 12; ++k) {
        int c = lane + 32 * k;
        float y = (v[k] - mean) * inv;
        if (g) y = y * g[c] + b[c];
        if (addvec) y += addvec[c];
        if (ada) y = y * (1.f + ada[(size_t)bb * astride + scoff + c]) +
                      ada[(size_t)bb * astride + shoff + c];
        yr[c] = (OutT)y;
    }
}

// concat(pos,fx,grid) -> LN over 20 -> bf16, padded to 32 cols (zeros)
__global__ void concat_ln_kernel(const float* __restrict__ pos, const float* __restrict__ fx,
                                 const float* __restrict__ grid, const float* __restrict__ g,
                                 const float* __restrict__ b, __bf16* __restrict__ out)
{
    int r = blockIdx.x * 256 + threadIdx.x;
    if (r >= B_ * N_) return;
    float v[IN_];
    v[0] = pos[(size_t)r * 2];  v[1] = pos[(size_t)r * 2 + 1];
    v[2] = fx[(size_t)r * 2];   v[3] = fx[(size_t)r * 2 + 1];
    #pragma unroll
    for (int i = 0; i < 16; ++i) v[4 + i] = grid[(size_t)r * 16 + i];
    float m = 0.f;
    #pragma unroll
    for (int i = 0; i < IN_; ++i) m += v[i];
    m *= (1.f / IN_);
    float var = 0.f;
    #pragma unroll
    for (int i = 0; i < IN_; ++i) { float d = v[i] - m; var += d * d; }
    var *= (1.f / IN_);
    float inv = rsqrtf(var + 1e-5f);
    __bf16* o = out + (size_t)r * 32;
    #pragma unroll
    for (int i = 0; i < IN_; ++i) o[i] = f2bf((v[i] - m) * inv * g[i] + b[i]);
    #pragma unroll
    for (int i = IN_; i < 32; ++i) o[i] = f2bf(0.f);
}

// s_in (B*M x 4) -> bf16 padded to 32 cols
__global__ void sin_bf_kernel(const float* __restrict__ s_in, __bf16* __restrict__ out)
{
    int idx = blockIdx.x * 256 + threadIdx.x;
    if (idx >= B_ * M_ * 32) return;
    int r = idx >> 5, k = idx & 31;
    out[idx] = f2bf((k < 4) ? s_in[(size_t)r * 4 + k] : 0.f);
}

// mu = t_embed(t) + mean_m( LN(s_in @ se2_w + se2_b) ); also sm = silu(mu)
__global__ __launch_bounds__(128)
void mu_kernel(const int* __restrict__ t, const float* __restrict__ s_in,
               const float* __restrict__ tw1, const float* __restrict__ tb1,
               const float* __restrict__ tw2, const float* __restrict__ tb2,
               const float* __restrict__ se2w, const float* __restrict__ se2b,
               const float* __restrict__ lng, const float* __restrict__ lnb,
               float* __restrict__ mu, float* __restrict__ sm)
{
    __shared__ float buf[96];
    __shared__ float buf2[96];
    const int bb = blockIdx.x;
    const int j = threadIdx.x;
    const float tv = (float)t[bb];
    if (j < 96) {
        int k = (j < 48) ? j : (j - 48);
        float fr = __expf(-9.210340371976184f * (float)k / 48.f);
        buf[j] = (j < 48) ? cosf(tv * fr) : sinf(tv * fr);
    }
    __syncthreads();
    if (j < 96) {
        float a = tb1[j];
        for (int i = 0; i < 96; ++i) a += buf[i] * tw1[i * 96 + j];
        buf2[j] = siluf_(a);
    }
    __syncthreads();
    float mut = 0.f;
    if (j < 96) {
        float a = tb2[j];
        for (int i = 0; i < 96; ++i) a += buf2[i] * tw2[i * 96 + j];
        mut = a;
    }
    float acc = 0.f;
    for (int m = 0; m < M_; ++m) {
        __syncthreads();
        if (j < 96) {
            const float* sr = s_in + ((size_t)bb * M_ + m) * 4;
            float a = se2b[j];
            #pragma unroll
            for (int i = 0; i < 4; ++i) a += sr[i] * se2w[i * 96 + j];
            buf[j] = a;
        }
        __syncthreads();
        if (j < 96) {
            float mean = 0.f;
            for (int i = 0; i < 96; ++i) mean += buf[i];
            mean *= (1.f / 96.f);
            float var = 0.f;
            for (int i = 0; i < 96; ++i) { float d = buf[i] - mean; var += d * d; }
            var *= (1.f / 96.f);
            float inv = rsqrtf(var + 1e-5f);
            acc += (buf[j] - mean) * inv * lng[j] + lnb[j];
        }
    }
    if (j < 96) {
        float mv = mut + acc * (1.f / M_);
        mu[bb * 96 + j] = mv;
        sm[bb * 96 + j] = siluf_(mv);
    }
}

// out[b, j] = bias[j] + sum_i sm[b,i] * W[i, C] (adaLN projections, tiny)
__global__ void matvec96_kernel(const float* __restrict__ sm, const float* __restrict__ W,
                                const float* __restrict__ bias, float* __restrict__ out, int C)
{
    int idx = blockIdx.x * 256 + threadIdx.x;
    if (idx >= B_ * C) return;
    int bb = idx / C, j = idx - bb * C;
    const float* s = sm + bb * 96;
    float a = bias[j];
    for (int i = 0; i < 96; ++i) a += s[i] * W[i * C + j];
    out[idx] = a;
}

// Cross attention: per (b, head, token) online softmax over M=256 keys, DH=8
__global__ __launch_bounds__(256)
void ca_attn_kernel(const float* __restrict__ Q, const float* __restrict__ KV,
                    __bf16* __restrict__ O)
{
    __shared__ float ks[M_][DH_];
    __shared__ float vs[M_][DH_];
    const int h = blockIdx.y, bb = blockIdx.z;
    const int tid = threadIdx.x;
    for (int e = tid; e < M_ * DH_; e += 256) {
        int m = e >> 3, d = e & 7;
        size_t base = ((size_t)bb * M_ + m) * (2 * CAI_);
        ks[m][d] = KV[base + h * DH_ + d];
        vs[m][d] = KV[base + CAI_ + h * DH_ + d];
    }
    __syncthreads();
    int n = blockIdx.x * 256 + tid;
    size_t qbase = ((size_t)bb * N_ + n) * CAI_ + h * DH_;
    float q[DH_];
    #pragma unroll
    for (int d = 0; d < DH_; ++d) q[d] = Q[qbase + d];
    float mx = -1e30f, l = 0.f;
    float acc[DH_] = {0.f,0.f,0.f,0.f,0.f,0.f,0.f,0.f};
    const float scale = 0.3535533905932738f;  // 1/sqrt(8)
    for (int m = 0; m < M_; ++m) {
        float s = 0.f;
        #pragma unroll
        for (int d = 0; d < DH_; ++d) s += q[d] * ks[m][d];
        s *= scale;
        float nm = fmaxf(mx, s);
        float corr = __expf(mx - nm);
        float p = __expf(s - nm);
        l = l * corr + p;
        #pragma unroll
        for (int d = 0; d < DH_; ++d) acc[d] = acc[d] * corr + p * vs[m][d];
        mx = nm;
    }
    float invl = 1.f / l;
    #pragma unroll
    for (int d = 0; d < DH_; ++d) O[qbase + d] = f2bf(acc[d] * invl);
}

// phys-attn slot weights: sw[b,n,h,:] = softmax( (x_mid_h @ pa_sw + pa_sb) / temp[h] )
__global__ void pa_softmax_kernel(const float* __restrict__ Xmid, const float* __restrict__ SWw,
                                  const float* __restrict__ SWb, const float* __restrict__ Temp,
                                  float* __restrict__ SW)
{
    int idx = blockIdx.x * 256 + threadIdx.x;
    if (idx >= B_ * N_ * HEADS_) return;
    int h = idx & 7;
    size_t bn = (size_t)(idx >> 3);
    float xm[DH_];
    #pragma unroll
    for (int d = 0; d < DH_; ++d) xm[d] = Xmid[bn * INNER_ + h * DH_ + d];
    float invt = 1.f / Temp[h];
    float lg[G_], mx = -1e30f;
    #pragma unroll
    for (int g = 0; g < G_; ++g) {
        float s = SWb[g];
        #pragma unroll
        for (int d = 0; d < DH_; ++d) s += xm[d] * SWw[d * G_ + g];
        s *= invt;
        lg[g] = s;
        mx = fmaxf(mx, s);
    }
    float sum = 0.f;
    #pragma unroll
    for (int g = 0; g < G_; ++g) { lg[g] = __expf(lg[g] - mx); sum += lg[g]; }
    float inv = 1.f / sum;
    #pragma unroll
    for (int g = 0; g < G_; ++g) SW[(size_t)idx * G_ + g] = lg[g] * inv;
}

// tok[b,h,g,c] = sum_n fx_mid * sw ; denom[b,h,g] = sum_n sw
__global__ __launch_bounds__(256)
void pa_tok_kernel(const float* __restrict__ FX, const float* __restrict__ SW,
                   float* __restrict__ TokRaw, float* __restrict__ Denom)
{
    const int bh = blockIdx.x;
    const int bb = bh >> 3, h = bh & 7;
    const int g = threadIdx.x >> 3, c = threadIdx.x & 7;
    float acc = 0.f, ds = 0.f;
    for (int n = 0; n < N_; ++n) {
        size_t bn = (size_t)bb * N_ + n;
        float s = SW[(bn * HEADS_ + h) * G_ + g];
        acc += s * FX[bn * INNER_ + h * DH_ + c];
        ds += s;
    }
    TokRaw[((size_t)bh * G_ + g) * DH_ + c] = acc;
    if (c == 0) Denom[bh * G_ + g] = ds;
}

// tiny inner SDPA on the G=32 slot tokens per (b,h)
__global__ __launch_bounds__(32)
void pa_inner_kernel(const float* __restrict__ TokRaw, const float* __restrict__ Denom,
                     const float* __restrict__ Wq, const float* __restrict__ Wk,
                     const float* __restrict__ Wv, float* __restrict__ O)
{
    __shared__ float qs[G_][DH_], ks[G_][DH_], vs[G_][DH_];
    const int bh = blockIdx.x;
    const int g = threadIdx.x;
    float tk[DH_];
    float dn = 1.f / (Denom[bh * G_ + g] + 1e-5f);
    #pragma unroll
    for (int c = 0; c < DH_; ++c) tk[c] = TokRaw[((size_t)bh * G_ + g) * DH_ + c] * dn;
    #pragma unroll
    for (int c = 0; c < DH_; ++c) {
        float q = 0.f, k = 0.f, v = 0.f;
        #pragma unroll
        for (int d = 0; d < DH_; ++d) {
            q += tk[d] * Wq[d * DH_ + c];
            k += tk[d] * Wk[d * DH_ + c];
            v += tk[d] * Wv[d * DH_ + c];
        }
        qs[g][c] = q; ks[g][c] = k; vs[g][c] = v;
    }
    __syncthreads();
    float sc[G_];
    float mx = -1e30f;
    #pragma unroll
    for (int gg = 0; gg < G_; ++gg) {
        float s = 0.f;
        #pragma unroll
        for (int d = 0; d < DH_; ++d) s += qs[g][d] * ks[gg][d];
        s *= 0.3535533905932738f;
        sc[gg] = s;
        mx = fmaxf(mx, s);
    }
    float sum = 0.f;
    #pragma unroll
    for (int gg = 0; gg < G_; ++gg) { sc[gg] = __expf(sc[gg] - mx); sum += sc[gg]; }
    float inv = 1.f / sum;
    #pragma unroll
    for (int c = 0; c < DH_; ++c) {
        float o = 0.f;
        #pragma unroll
        for (int gg = 0; gg < G_; ++gg) o += sc[gg] * vs[gg][c];
        O[((size_t)bh * G_ + g) * DH_ + c] = o * inv;
    }
}

// out[b,n,h,c] = sum_g o[b,h,g,c] * sw[b,n,h,g]   (bf16 out -> next GEMM)
__global__ __launch_bounds__(256)
void pa_scatter_kernel(const float* __restrict__ O, const float* __restrict__ SW,
                       __bf16* __restrict__ Out)
{
    __shared__ float os[G_][DH_];
    const int h = blockIdx.y, bb = blockIdx.z;
    const int tid = threadIdx.x;
    if (tid < G_ * DH_)
        os[tid >> 3][tid & 7] = O[((size_t)(bb * HEADS_ + h) * G_) * DH_ + tid];
    __syncthreads();
    int n = blockIdx.x * 256 + tid;
    size_t bn = (size_t)bb * N_ + n;
    const float* swp = SW + (bn * HEADS_ + h) * G_;
    float acc[DH_] = {0.f,0.f,0.f,0.f,0.f,0.f,0.f,0.f};
    for (int g = 0; g < G_; ++g) {
        float s = swp[g];
        #pragma unroll
        for (int c = 0; c < DH_; ++c) acc[c] += s * os[g][c];
    }
    #pragma unroll
    for (int c = 0; c < DH_; ++c) Out[bn * INNER_ + h * DH_ + c] = f2bf(acc[c]);
}

// final projection H -> 1 (reads bf16 hidden)
__global__ void final_out_kernel(const __bf16* __restrict__ Hid, const float* __restrict__ W2,
                                 const float* __restrict__ b2, float* __restrict__ Out)
{
    int r = blockIdx.x * 256 + threadIdx.x;
    if (r >= B_ * N_) return;
    const __bf16* hr = Hid + (size_t)r * H_;
    float a = b2[0];
    for (int c = 0; c < H_; ++c) a += (float)hr[c] * W2[c];
    Out[r] = a;
}

// ---------------------------------------------------------------------------
extern "C" void kernel_launch(void* const* d_in, const int* in_sizes, int n_in,
                              void* d_out, int out_size, void* d_ws, size_t ws_size,
                              hipStream_t stream)
{
    (void)in_sizes; (void)n_in; (void)out_size; (void)ws_size;
    auto P = [&](int i) { return (const float*)d_in[i]; };

    const float* pos  = P(0);
    const float* fx   = P(1);
    const float* grid = P(2);
    const int*   t    = (const int*)d_in[3];
    const float* s_in = P(4);
    const float* pre_ln0_g = P(5),  *pre_ln0_b = P(6);
    const float* pre_w = P(7),      *pre_b = P(8);
    const float* pre_ln1_g = P(9),  *pre_ln1_b = P(10);
    const float* placeholder = P(11);
    const float* se_w = P(12), *se_b = P(13), *se_ln_g = P(14), *se_ln_b = P(15);
    const float* se2_w = P(16), *se2_b = P(17), *se2_ln_g = P(18), *se2_ln_b = P(19);
    const float* t_w1 = P(20), *t_b1 = P(21), *t_w2 = P(22), *t_b2 = P(23);
    const float* fin_ada_w = P(24), *fin_ada_b = P(25);
    const float* fin_w1 = P(26), *fin_b1 = P(27), *fin_w2 = P(28), *fin_b2 = P(29);
    const int LBASE = 30, NL = 36;

    const size_t BN  = (size_t)B_ * N_;
    const size_t BNH = BN * H_;
    const size_t BMH = (size_t)B_ * M_ * H_;

    // ---- f32 workspace ----
    float* ws   = (float*)d_ws;
    float* X    = ws;                               // B*N*H
    float* BUF1 = ws + BNH;                         // B*N*H (pre-out / Qb / FX/Xmid)
    float* SWb_ = ws + 2 * BNH;                     // B*N*HEADS*G
    float* Semb = SWb_ + BN * HEADS_ * G_;          // B*M*H
    float* KVb  = Semb + BMH;                       // B*M*64
    float* MUb  = KVb + (size_t)B_ * M_ * 64;       // B*MU
    float* SMb  = MUb + B_ * MU_;
    float* A1   = SMb + B_ * MU_;                   // B*6H
    float* A2   = A1 + B_ * 6 * H_;
    float* AF   = A2 + B_ * 6 * H_;                 // B*2H
    float* TOKR = AF + B_ * 2 * H_;                 // B*HEADS*G*DH
    float* DEN  = TOKR + B_ * HEADS_ * G_ * DH_;    // B*HEADS*G
    float* Ob   = DEN + B_ * HEADS_ * G_;           // B*HEADS*G*DH
    float* f32end = Ob + B_ * HEADS_ * G_ * DH_;

    // ---- bf16 workspace ----
    __bf16* bf    = (__bf16*)f32end;
    __bf16* XMODb = bf;                 bf += BNH;          // B*N*H
    __bf16* FFHb  = bf;                 bf += BNH;          // B*N*H
    __bf16* XCATb = bf;                 bf += BN * 32;      // padded 20->32
    __bf16* SINb  = bf;                 bf += (size_t)B_ * M_ * 32;
    __bf16* SembB = bf;                 bf += BMH;
    __bf16* AOutB = bf;                 bf += BN * CAI_;
    __bf16* PAoutB= bf;                 bf += BN * INNER_;
    __bf16* wpool = bf;                                     // weight pool (bump)

    float* Qb    = BUF1;                    // B*N*CAI (f32, attn input)
    float* FXmid = BUF1;                    // B*N*64
    float* Xmid  = BUF1 + BN * INNER_;      // B*N*64
    float* FFHf  = BUF1;                    // pre-GEMM f32 out

    // convert+transpose+pad one weight into the pool
    auto conv = [&](const float* W, int K, int C) -> const __bf16* {
        int Ka = (K + 31) & ~31;
        int Ca = (C + 63) & ~63;
        __bf16* dst = wpool;
        int n = Ka * Ca;
        wpool += n;
        wconvT_kernel<<<dim3((n + 255) / 256), dim3(256), 0, stream>>>(W, dst, K, C, Ka, n);
        return dst;
    };
    auto gemm = [&](const __bf16* A, const __bf16* Wt, const float* bias,
                    float* DstF, __bf16* DstB, const float* Resid, const float* Gate,
                    int rows, int K, int C, int epi, int gstride, int goff, int rowsPerB) {
        int Ka = (K + 31) & ~31;
        dim3 gr(rows / 64, (C + 63) / 64);
        gemm_bf16<<<gr, dim3(128), 0, stream>>>(A, Wt, bias, DstF, DstB, Resid, Gate,
                                                Ka, C, epi, gstride, goff, rowsPerB);
    };
    auto lnf = [&](const float* Xi, float* Yo, int rows, const float* g, const float* b,
                   const float* addvec, float eps) {
        lnmod_kernel<float><<<dim3((rows + 7) / 8), dim3(256), 0, stream>>>(
            Xi, Yo, rows, g, b, nullptr, 0, 0, 0, addvec, N_, eps);
    };
    auto lnb = [&](const float* Xi, __bf16* Yo, int rows, const float* g, const float* b,
                   const float* ada, int astride, int shoff, int scoff, float eps) {
        lnmod_kernel<__bf16><<<dim3((rows + 7) / 8), dim3(256), 0, stream>>>(
            Xi, Yo, rows, g, b, ada, astride, shoff, scoff, nullptr, N_, eps);
    };

    // --- preprocessing ---
    concat_ln_kernel<<<dim3((unsigned)((BN + 255) / 256)), dim3(256), 0, stream>>>(
        pos, fx, grid, pre_ln0_g, pre_ln0_b, XCATb);
    const __bf16* wt_pre = conv(pre_w, IN_, H_);
    gemm(XCATb, wt_pre, pre_b, FFHf, nullptr, nullptr, nullptr, (int)BN, IN_, H_, 0, 0, 0, N_);
    lnf(FFHf, X, (int)BN, pre_ln1_g, pre_ln1_b, placeholder, 1e-5f);

    sin_bf_kernel<<<dim3((B_ * M_ * 32 + 255) / 256), dim3(256), 0, stream>>>(s_in, SINb);
    const __bf16* wt_se = conv(se_w, 4, H_);
    gemm(SINb, wt_se, se_b, Semb, nullptr, nullptr, nullptr, B_ * M_, 4, H_, 0, 0, 0, N_);
    lnb(Semb, SembB, B_ * M_, se_ln_g, se_ln_b, nullptr, 0, 0, 0, 1e-5f);

    mu_kernel<<<dim3(B_), dim3(128), 0, stream>>>(t, s_in, t_w1, t_b1, t_w2, t_b2,
                                                  se2_w, se2_b, se2_ln_g, se2_ln_b, MUb, SMb);

    // --- layers ---
    for (int L = 0; L < DEPTH_; ++L) {
        const int pb = LBASE + L * NL;
        const float *ca_q = P(pb+0), *ca_kv = P(pb+1), *ca_ow = P(pb+2), *ca_ob = P(pb+3);
        const float *f1_w1 = P(pb+4), *f1_b1 = P(pb+5), *f1_w2 = P(pb+6), *f1_b2 = P(pb+7);
        const float *pa_fxw = P(pb+8), *pa_fxb = P(pb+9), *pa_xw = P(pb+10), *pa_xb = P(pb+11);
        const float *pa_sw = P(pb+12), *pa_sb = P(pb+13), *pa_temp = P(pb+14);
        const float *pa_q = P(pb+15), *pa_k = P(pb+16), *pa_v = P(pb+17);
        const float *pa_ow = P(pb+18), *pa_ob = P(pb+19);
        const float *f2_w1 = P(pb+20), *f2_b1 = P(pb+21), *f2_w2 = P(pb+22), *f2_b2 = P(pb+23);
        const float *ada1_w = P(pb+24), *ada1_b = P(pb+25), *ada2_w = P(pb+26), *ada2_b = P(pb+27);
        const float *n1_g = P(pb+28), *n1_b = P(pb+29), *n2_g = P(pb+30), *n2_b = P(pb+31);
        const float *n3_g = P(pb+32), *n3_b = P(pb+33), *n4_g = P(pb+34), *n4_b = P(pb+35);

        matvec96_kernel<<<dim3((B_ * 6 * H_ + 255) / 256), dim3(256), 0, stream>>>(
            SMb, ada1_w, ada1_b, A1, 6 * H_);

        // cross attention block
        lnb(X, XMODb, (int)BN, n1_g, n1_b, A1, 6 * H_, 0, H_, 1e-5f);
        const __bf16* wcaq  = conv(ca_q, H_, CAI_);
        const __bf16* wcakv = conv(ca_kv, H_, 2 * CAI_);
        gemm(XMODb, wcaq, nullptr, Qb, nullptr, nullptr, nullptr, (int)BN, H_, CAI_, 0, 0, 0, N_);
        gemm(SembB, wcakv, nullptr, KVb, nullptr, nullptr, nullptr, B_ * M_, H_, 2 * CAI_, 0, 0, 0, N_);
        ca_attn_kernel<<<dim3(N_ / 256, CAH_, B_), dim3(256), 0, stream>>>(Qb, KVb, AOutB);
        const __bf16* wcaow = conv(ca_ow, CAI_, H_);
        gemm(AOutB, wcaow, ca_ob, X, nullptr, X, A1, (int)BN, CAI_, H_, 2, 6 * H_, 2 * H_, N_);

        // FF1
        lnb(X, XMODb, (int)BN, n2_g, n2_b, A1, 6 * H_, 3 * H_, 4 * H_, 1e-5f);
        const __bf16* wf11 = conv(f1_w1, H_, H_);
        const __bf16* wf12 = conv(f1_w2, H_, H_);
        gemm(XMODb, wf11, f1_b1, nullptr, FFHb, nullptr, nullptr, (int)BN, H_, H_, 1, 0, 0, N_);
        gemm(FFHb, wf12, f1_b2, X, nullptr, X, A1, (int)BN, H_, H_, 2, 6 * H_, 5 * H_, N_);

        // phys attention block
        matvec96_kernel<<<dim3((B_ * 6 * H_ + 255) / 256), dim3(256), 0, stream>>>(
            SMb, ada2_w, ada2_b, A2, 6 * H_);
        lnb(X, XMODb, (int)BN, n3_g, n3_b, A2, 6 * H_, 0, H_, 1e-5f);
        const __bf16* wfxw = conv(pa_fxw, H_, INNER_);
        const __bf16* wxw  = conv(pa_xw, H_, INNER_);
        gemm(XMODb, wfxw, pa_fxb, FXmid, nullptr, nullptr, nullptr, (int)BN, H_, INNER_, 0, 0, 0, N_);
        gemm(XMODb, wxw,  pa_xb,  Xmid,  nullptr, nullptr, nullptr, (int)BN, H_, INNER_, 0, 0, 0, N_);
        pa_softmax_kernel<<<dim3(B_ * N_ * HEADS_ / 256), dim3(256), 0, stream>>>(
            Xmid, pa_sw, pa_sb, pa_temp, SWb_);
        pa_tok_kernel<<<dim3(B_ * HEADS_), dim3(256), 0, stream>>>(FXmid, SWb_, TOKR, DEN);
        pa_inner_kernel<<<dim3(B_ * HEADS_), dim3(32), 0, stream>>>(TOKR, DEN, pa_q, pa_k, pa_v, Ob);
        pa_scatter_kernel<<<dim3(N_ / 256, HEADS_, B_), dim3(256), 0, stream>>>(Ob, SWb_, PAoutB);
        const __bf16* wpaow = conv(pa_ow, INNER_, H_);
        gemm(PAoutB, wpaow, pa_ob, X, nullptr, X, A2, (int)BN, INNER_, H_, 2, 6 * H_, 2 * H_, N_);

        // FF2
        lnb(X, XMODb, (int)BN, n4_g, n4_b, A2, 6 * H_, 3 * H_, 4 * H_, 1e-5f);
        const __bf16* wf21 = conv(f2_w1, H_, H_);
        const __bf16* wf22 = conv(f2_w2, H_, H_);
        gemm(XMODb, wf21, f2_b1, nullptr, FFHb, nullptr, nullptr, (int)BN, H_, H_, 1, 0, 0, N_);
        gemm(FFHb, wf22, f2_b2, X, nullptr, X, A2, (int)BN, H_, H_, 2, 6 * H_, 5 * H_, N_);
    }

    // --- final head ---
    matvec96_kernel<<<dim3((B_ * 2 * H_ + 255) / 256), dim3(256), 0, stream>>>(
        SMb, fin_ada_w, fin_ada_b, AF, 2 * H_);
    lnb(X, XMODb, (int)BN, nullptr, nullptr, AF, 2 * H_, 0, H_, 1e-6f);
    const __bf16* wfin1 = conv(fin_w1, H_, H_);
    gemm(XMODb, wfin1, fin_b1, nullptr, FFHb, nullptr, nullptr, (int)BN, H_, H_, 1, 0, 0, N_);
    final_out_kernel<<<dim3((unsigned)((BN + 255) / 256)), dim3(256), 0, stream>>>(
        FFHb, fin_w2, fin_b2, (float*)d_out);
}
